// Pointnet2MSG_60421599920368
// MI455X (gfx1250) — matrix operand order repaired and checked by reading, not compiled, not run on hardware
//
#include <hip/hip_runtime.h>
#include <cstdint>
#include <cstddef>

// ---------------------------------------------------------------------------
// MI455X (gfx1250) implementation of the fused point/image cross-attention
// block. All six GEMMs run on v_wmma_f32_16x16x32_f16 (f16 in, f32 accum).
// Roofline: ~51 GFLOP vs ~0.6 GB HBM traffic -> compute/memory balanced at
// f16 WMMA rates; everything else (gather, softmax, BN) is bandwidth-bound
// and kept coalesced.
// ---------------------------------------------------------------------------

typedef __attribute__((ext_vector_type(16))) _Float16 v16h;
typedef __attribute__((ext_vector_type(8)))  _Float16 v8h;
typedef __attribute__((ext_vector_type(8)))  float    v8f;
typedef __attribute__((ext_vector_type(4)))  float    v4f;

#define BN_INV   0.9999950000374997f   // 1/sqrt(1 + 1e-5)
#define QKV_RSQ  0.08838834764831845f  // 1/sqrt(128)

#define NB    8
#define IC    256
#define PCH   512
#define QKV   128
#define NPTS  16384
#define HWPIX (96 * 312)

// D = A(16x32 f16) * B(32x16 f16) + C(16x16 f32)
static __device__ __forceinline__ v8f wmma32(v16h a, v16h b, v8f c) {
  return __builtin_amdgcn_wmma_f32_16x16x32_f16(false, a, false, b,
                                                (short)0, c, false, false);
}

// Within-wave LDS handoff: DS ops are issued in order per wave (wave32 is
// lockstep), s_wait_dscnt 0 guarantees completion; "memory" stops reordering.
static __device__ __forceinline__ void lds_fence() {
  asm volatile("s_wait_dscnt 0x0" ::: "memory");
}

// A fragment (16x32 f16, row-major source, row stride ld halves).
// Lane layout (ISA 7.12.2): lanes 0-15 = M, lane-group selects K interleave:
// halves 0-7 -> K = k0 + lg*8 + i ; halves 8-15 -> K = k0 + 16 + lg*8 + i.
static __device__ __forceinline__ v16h a_frag(const _Float16* base, int ld,
                                              int k0, int lane) {
  const int m  = lane & 15;
  const int lg = (lane >> 4) & 1;
  const v8h lo = *(const v8h*)(base + m * ld + k0 + lg * 8);
  const v8h hi = *(const v8h*)(base + m * ld + k0 + 16 + lg * 8);
  v16h r;
#pragma unroll
  for (int i = 0; i < 8; ++i) { r[i] = lo[i]; r[8 + i] = hi[i]; }
  return r;
}

// B fragment (32x16 f16) from a column-major f16 weight table Wt[col][k]
// (col stride = K). Lanes 0-15 = N; halves 0-7 -> K = k0 + lg*8 + i,
// halves 8-15 -> K = k0 + 16 + lg*8 + i -> two 16B loads per lane.
static __device__ __forceinline__ v16h b_frag(const _Float16* Wt, int K,
                                              int colBase, int k0, int lane) {
  const int c  = colBase + (lane & 15);
  const int kk = k0 + ((lane >> 4) & 1) * 8;
  const v8h* p = (const v8h*)(Wt + (size_t)c * K + kk);
  const v8h lo = p[0];
  const v8h hi = p[2];
  v16h r;
#pragma unroll
  for (int i = 0; i < 8; ++i) { r[i] = lo[i]; r[8 + i] = hi[i]; }
  return r;
}

// --------------------------- K0: weight prep -------------------------------
// dst[c*R + r] = f16(src[r*C + c])  (column-major f16 table)
__global__ void k0_transpose_f16(const float* __restrict__ src,
                                 _Float16* __restrict__ dst, int R, int C) {
  int i = blockIdx.x * blockDim.x + threadIdx.x;
  if (i < R * C) {
    int r = i / C, c = i - r * C;
    dst[c * R + r] = (_Float16)src[i];
  }
}

// --------------------------- K1: gather + bni ------------------------------
__global__ void k1_gather_bn(const float* __restrict__ img,
                             const int* __restrict__ li,
                             const float* __restrict__ g,
                             const float* __restrict__ b,
                             float* __restrict__ imgff) {
  const size_t total = (size_t)NB * NPTS * IC;
  for (size_t i = (size_t)blockIdx.x * blockDim.x + threadIdx.x; i < total;
       i += (size_t)gridDim.x * blockDim.x) {
    int c     = (int)(i & (IC - 1));
    size_t bn = i >> 8;                 // b*NPTS + n
    int bb    = (int)(bn >> 14);        // NPTS = 2^14
    int p     = li[bn];
    float x   = img[((size_t)bb * IC + c) * HWPIX + p];
    imgff[i]  = g[c] * BN_INV * x + b[c];
  }
}

// ------------------- K2: q/k/v GEMMs + affinity ----------------------------
// One 16-point tile per workgroup; 4 waves; wave w owns qkv column tiles
// {2w, 2w+1}. q.k row-dots reduced with ds_add_f32 LDS atomics.
__global__ void __launch_bounds__(128)
k2_qkv_affinity(const float* __restrict__ pts, const float* __restrict__ imgff,
                const _Float16* __restrict__ qWt, const _Float16* __restrict__ kWt,
                const _Float16* __restrict__ vWt,
                const float* __restrict__ qb, const float* __restrict__ kb,
                const float* __restrict__ vb,
                const float* __restrict__ bpg, const float* __restrict__ bpb,
                _Float16* __restrict__ vout, float* __restrict__ aff) {
  __shared__ __align__(16) _Float16 sPf[16 * PCH];  // 16 KB
  __shared__ __align__(16) _Float16 sIm[16 * IC];   //  8 KB
  __shared__ float sAff[16];

  const int tile = blockIdx.x;            // 0..8191
  const int b    = tile >> 10;            // 1024 tiles per batch
  const int n0   = (tile & 1023) << 4;
  const int tid  = threadIdx.x;

  if (tid == 0) {                         // warm weight lines (global_prefetch)
    __builtin_prefetch(qWt, 0, 3);
    __builtin_prefetch(kWt, 0, 3);
    __builtin_prefetch(vWt, 0, 3);
  }

  // stage bnp(point_feas^T) tile: 16 x 512 f16 (global reads: 64B runs)
  for (int e = tid; e < PCH * 16; e += 128) {
    int c = e >> 4, n = e & 15;
    float x = pts[((size_t)b * PCH + c) * NPTS + n0 + n];
    sPf[n * PCH + c] = (_Float16)(x * bpg[c] * BN_INV + bpb[c]);
  }
  // stage img_ff tile: 16 x 256 f16 (fully coalesced)
  for (int e = tid; e < IC * 16; e += 128) {
    int n = e >> 8, c = e & 255;
    sIm[n * IC + c] = (_Float16)imgff[((size_t)b * NPTS + n0 + n) * IC + c];
  }
  if (tid < 16) sAff[tid] = 0.0f;
  __syncthreads();

  const int wave = tid >> 5, lane = tid & 31;
  const int mg = ((lane >> 4) & 1) * 8;

#pragma unroll
  for (int jj = 0; jj < 2; ++jj) {
    const int j = wave * 2 + jj;          // qkv column tile 0..7
    v8f qa = {}, ka = {}, va = {};
#pragma unroll
    for (int kc = 0; kc < 16; ++kc)       // q: K = 512
      qa = wmma32(a_frag(sPf, PCH, kc * 32, lane),
                  b_frag(qWt, PCH, j * 16, kc * 32, lane), qa);
#pragma unroll
    for (int kc = 0; kc < 8; ++kc) {      // k,v: K = 256
      v16h a = a_frag(sIm, IC, kc * 32, lane);
      ka = wmma32(a, b_frag(kWt, IC, j * 16, kc * 32, lane), ka);
      va = wmma32(a, b_frag(vWt, IC, j * 16, kc * 32, lane), va);
    }
    const int c = j * 16 + (lane & 15);
    const float qbi = qb[c], kbi = kb[c], vbi = vb[c];
#pragma unroll
    for (int r = 0; r < 8; ++r) {
      const int m = mg + r;
      float qv = qa[r] + qbi;
      float kv = ka[r] + kbi;
      float vv = va[r] + vbi;
      vout[((size_t)b * NPTS + n0 + m) * QKV + c] = (_Float16)vv;
      atomicAdd(&sAff[m], qv * kv);       // ds_add_f32
    }
  }
  __syncthreads();
  if (tid < 16) aff[(size_t)b * NPTS + n0 + tid] = sAff[tid] * QKV_RSQ;
}

// --------------------------- K3: softmax over N ----------------------------
__global__ void __launch_bounds__(256) k3_softmax(float* __restrict__ aff) {
  __shared__ float red[256];
  const int b = blockIdx.x, tid = threadIdx.x;
  float* a = aff + (size_t)b * NPTS;
  float mx = -3.4e38f;
  for (int i = tid; i < NPTS; i += 256) mx = fmaxf(mx, a[i]);
  red[tid] = mx;
  __syncthreads();
  for (int s = 128; s > 0; s >>= 1) {
    if (tid < s) red[tid] = fmaxf(red[tid], red[tid + s]);
    __syncthreads();
  }
  mx = red[0];
  __syncthreads();
  float sm = 0.0f;
  for (int i = tid; i < NPTS; i += 256) sm += __expf(a[i] - mx);
  red[tid] = sm;
  __syncthreads();
  for (int s = 128; s > 0; s >>= 1) {
    if (tid < s) red[tid] += red[tid + s];
    __syncthreads();
  }
  const float inv = 1.0f / red[0];
  for (int i = tid; i < NPTS; i += 256) a[i] = __expf(a[i] - mx) * inv;
}

// ---------- K4: (w*v)@oW + res + bn ; FFN ; final bn ; transpose out -------
// 4 waves/block, 2 tiles/wave. oan lives in 16 x v8f accumulators; per-wave
// LDS scratch round-trips build A fragments for fc1/fc2 and the 16x16 output
// transpose so (B,ic,N) stores are 32B-contiguous.
__global__ void __launch_bounds__(128)
k4_out_ffn(const float* __restrict__ imgff, const _Float16* __restrict__ v16,
           const float* __restrict__ wts, const _Float16* __restrict__ oWt,
           const float* __restrict__ ob, const _Float16* __restrict__ f1t,
           const float* __restrict__ f1b, const _Float16* __restrict__ f2t,
           const float* __restrict__ f2b, const float* __restrict__ big,
           const float* __restrict__ bib, float* __restrict__ out) {
  __shared__ __align__(16) _Float16 sOan[4][16 * 264];  // per-wave 16x256 (ld 264)
  __shared__ __align__(16) _Float16 sT[4][16 * 72];     // per-wave 16x64  (ld 72)

  const int tid  = threadIdx.x;
  const int wave = tid >> 5, lane = tid & 31;
  const int m    = lane & 15;
  const int lg   = (lane >> 4) & 1;
  _Float16* so = sOan[wave];
  _Float16* st = sT[wave];
  float*   tsp = (float*)sOan[wave];                    // 16x16 f32 view

  if (tid == 0) {
    __builtin_prefetch(oWt, 0, 3);
    __builtin_prefetch(f1t, 0, 3);
    __builtin_prefetch(f2t, 0, 3);
  }

  for (int t = 0; t < 2; ++t) {
    const int tile = blockIdx.x * 8 + wave * 2 + t;     // 0..8191
    const int b    = tile >> 10;
    const int n0   = (tile & 1023) << 4;
    const size_t rowBase = (size_t)b * NPTS + n0;

    // A fragments of softmax-weighted v, straight from global (16B loads)
    const _Float16 hw = (_Float16)wts[rowBase + m];
    v16h A1[4];
#pragma unroll
    for (int kc = 0; kc < 4; ++kc) {
      v16h a = a_frag(v16 + rowBase * QKV, QKV, kc * 32, lane);
#pragma unroll
      for (int i = 0; i < 16; ++i) a[i] = a[i] * hw;
      A1[kc] = a;
    }

    // ro = (w*v) @ oW   (16 col tiles x 4 K-chunks = 64 WMMA)
    v8f acc[16];
#pragma unroll
    for (int j = 0; j < 16; ++j) {
      v8f c = {};
#pragma unroll
      for (int kc = 0; kc < 4; ++kc)
        c = wmma32(A1[kc], b_frag(oWt, QKV, j * 16, kc * 32, lane), c);
      acc[j] = c;
    }

    // oan = bni(ro + o_b + img_ff); keep f32 in acc, f16 copy to LDS
    lds_fence();  // prior-iteration LDS reads done before overwrite
#pragma unroll
    for (int j = 0; j < 16; ++j) {
      const int c  = j * 16 + m;
      const float obi = ob[c], gi = big[c] * BN_INV, bi = bib[c];
#pragma unroll
      for (int r = 0; r < 8; ++r) {
        const int mm = lg * 8 + r;
        float x = acc[j][r] + obi + imgff[(rowBase + mm) * IC + c];
        float o = gi * x + bi;
        acc[j][r] = o;
        so[mm * 264 + c] = (_Float16)o;
      }
    }
    lds_fence();

    // fc1 + relu -> sT   (4 col tiles x 8 K-chunks = 32 WMMA)
#pragma unroll
    for (int j = 0; j < 4; ++j) {
      v8f c = {};
#pragma unroll
      for (int kc = 0; kc < 8; ++kc)
        c = wmma32(a_frag(so, 264, kc * 32, lane),
                   b_frag(f1t, IC, j * 16, kc * 32, lane), c);
      const int cc = j * 16 + m;
      const float fb = f1b[cc];
#pragma unroll
      for (int r = 0; r < 8; ++r) {
        float x = c[r] + fb;
        st[(lg * 8 + r) * 72 + cc] = (_Float16)(x > 0.0f ? x : 0.0f);
      }
    }
    lds_fence();

    v16h A3[2];
    A3[0] = a_frag(st, 72, 0, lane);
    A3[1] = a_frag(st, 72, 32, lane);

    // fc2 seeded with oan accumulators; final bn; transpose; coalesced store
#pragma unroll
    for (int j = 0; j < 16; ++j) {
      v8f c = acc[j];                                   // oan + (t@fc2)
      c = wmma32(A3[0], b_frag(f2t, 64, j * 16, 0, lane), c);
      c = wmma32(A3[1], b_frag(f2t, 64, j * 16, 32, lane), c);
      const int cc = j * 16 + m;
      const float gi = big[cc] * BN_INV, bi = bib[cc], fb = f2b[cc];
      lds_fence();                                      // prior tsp reads done
#pragma unroll
      for (int r = 0; r < 8; ++r)
        tsp[m * 16 + lg * 8 + r] = gi * (c[r] + fb) + bi;  // [c_local][n_local]
      lds_fence();
      const int cl = lane >> 1, hf = lane & 1;          // 2 lanes per channel
      v4f d0 = *(const v4f*)(tsp + cl * 16 + hf * 8);
      v4f d1 = *(const v4f*)(tsp + cl * 16 + hf * 8 + 4);
      float* po = out + ((size_t)b * IC + j * 16 + cl) * NPTS + n0 + hf * 8;
      *(v4f*)po       = d0;
      *(v4f*)(po + 4) = d1;
    }
  }
}

// ---------------------------------------------------------------------------
extern "C" void kernel_launch(void* const* d_in, const int* in_sizes, int n_in,
                              void* d_out, int out_size, void* d_ws,
                              size_t ws_size, hipStream_t stream) {
  (void)in_sizes; (void)n_in; (void)out_size; (void)ws_size;
  const float* img = (const float*)d_in[0];
  const float* pts = (const float*)d_in[1];
  const int*   li  = (const int*)d_in[2];
  const float* qW  = (const float*)d_in[3];  const float* qb  = (const float*)d_in[4];
  const float* kW  = (const float*)d_in[5];  const float* kb  = (const float*)d_in[6];
  const float* vW  = (const float*)d_in[7];  const float* vb  = (const float*)d_in[8];
  const float* oW  = (const float*)d_in[9];  const float* ob  = (const float*)d_in[10];
  const float* big = (const float*)d_in[11]; const float* bib = (const float*)d_in[12];
  const float* bpg = (const float*)d_in[13]; const float* bpb = (const float*)d_in[14];
  const float* f1W = (const float*)d_in[15]; const float* f1b = (const float*)d_in[16];
  const float* f2W = (const float*)d_in[17]; const float* f2b = (const float*)d_in[18];
  float* out = (float*)d_out;

  // workspace layout (bytes)
  char* ws = (char*)d_ws;
  float*    imgff = (float*)ws;                          // 8*16384*256*4 = 134217728
  _Float16* v16   = (_Float16*)(ws + 134217728);         // 8*16384*128*2 =  33554432
  float*    aff   = (float*)(ws + 167772160);            // 8*16384*4     =    524288
  _Float16* qWt   = (_Float16*)(ws + 168296448);         // 128x512 f16
  _Float16* kWt   = (_Float16*)(ws + 168427520);         // 128x256 f16
  _Float16* vWt   = (_Float16*)(ws + 168493056);         // 128x256 f16
  _Float16* oWt   = (_Float16*)(ws + 168558592);         // 256x128 f16
  _Float16* f1t   = (_Float16*)(ws + 168624128);         //  64x256 f16
  _Float16* f2t   = (_Float16*)(ws + 168656896);         // 256x 64 f16

  // K0: f16 column-major weight tables
  k0_transpose_f16<<<(PCH * QKV + 255) / 256, 256, 0, stream>>>(qW, qWt, PCH, QKV);
  k0_transpose_f16<<<(IC * QKV + 255) / 256, 256, 0, stream>>>(kW, kWt, IC, QKV);
  k0_transpose_f16<<<(IC * QKV + 255) / 256, 256, 0, stream>>>(vW, vWt, IC, QKV);
  k0_transpose_f16<<<(QKV * IC + 255) / 256, 256, 0, stream>>>(oW, oWt, QKV, IC);
  k0_transpose_f16<<<(IC * 64 + 255) / 256, 256, 0, stream>>>(f1W, f1t, IC, 64);
  k0_transpose_f16<<<(64 * IC + 255) / 256, 256, 0, stream>>>(f2W, f2t, 64, IC);

  // K1: gather + bni
  k1_gather_bn<<<131072, 256, 0, stream>>>(img, li, big, bib, imgff);

  // K2: q/k/v + affinity (one 16-point tile per WG)
  k2_qkv_affinity<<<NB * (NPTS / 16), 128, 0, stream>>>(
      pts, imgff, qWt, kWt, vWt, qb, kb, vb, bpg, bpb, v16, aff);

  // K3: softmax over N per batch
  k3_softmax<<<NB, 256, 0, stream>>>(aff);

  // K4: attention output + FFN + final BN + transposed store
  k4_out_ffn<<<NB * (NPTS / 16) / 8, 128, 0, stream>>>(
      imgff, v16, aff, oWt, ob, f1t, f1b, f2t, f2b, big, bib, out);
}
